// BitConvNeXtBlock_35605278884130
// MI455X (gfx1250) — 8-bit matrix operand order repaired and checked by reading, compiled, not run on hardware
//
#include <hip/hip_runtime.h>
#include <hip/hip_bf16.h>
#include <stdint.h>


typedef __attribute__((ext_vector_type(8))) int v8i;

#define BDIM 8
#define CDIM 512
#define TDIM 4096
#define IDIM 2048
#define NTOK (BDIM * TDIM)

__device__ __forceinline__ void async_tile_load(const int8_t* __restrict__ gsrc,
                                                int8_t* lds_dst, int bytes,
                                                int tid, int nthreads) {
    const uint32_t lbase = (uint32_t)(uintptr_t)lds_dst;
    for (int off = tid * 16; off < bytes; off += nthreads * 16) {
        const uint64_t ga = (uint64_t)(uintptr_t)(gsrc + off);
        const uint32_t la = lbase + (uint32_t)off;
        asm volatile("global_load_async_to_lds_b128 %0, %1, off"
                     :: "v"(la), "v"(ga) : "memory");
    }
}

__device__ __forceinline__ void async_wait_all() {
    asm volatile("s_wait_asynccnt 0x0" ::: "memory");
}

__global__ void init_scal_kernel(float* scal) {
    if (threadIdx.x < 2) scal[threadIdx.x] = 0.0f;
}

__global__ __launch_bounds__(256) void absmean_kernel(const float* __restrict__ w,
                                                      int n, float* __restrict__ out) {
    __shared__ float red[256];
    float acc = 0.0f;
    for (int i = blockIdx.x * 256 + threadIdx.x; i < n; i += gridDim.x * 256)
        acc += fabsf(w[i]);
    red[threadIdx.x] = acc;
    __syncthreads();
    for (int s = 128; s > 0; s >>= 1) {
        if (threadIdx.x < s) red[threadIdx.x] += red[threadIdx.x + s];
        __syncthreads();
    }
    if (threadIdx.x == 0) atomicAdd(out, red[0]);
}

__global__ __launch_bounds__(256) void wquant_kernel(const float* __restrict__ w, int n,
                                                     const float* __restrict__ sumabs,
                                                     int8_t* __restrict__ q) {
    const float mean = fmaxf(*sumabs * (1.0f / (float)(IDIM * CDIM)), 1e-5f);
    const float s = 1.0f / mean;
    for (int i = blockIdx.x * 256 + threadIdx.x; i < n; i += gridDim.x * 256) {
        float v = rintf(w[i] * s);
        v = fminf(fmaxf(v, -1.0f), 1.0f);
        q[i] = (int8_t)v;
    }
}

__global__ __launch_bounds__(256) void conv_norm_quant_kernel(
    const float* __restrict__ x, const float* __restrict__ dww,
    const float* __restrict__ dwb, const float* __restrict__ nw,
    int8_t* __restrict__ a1, float* __restrict__ inv_sa1) {
    const int wave = threadIdx.x >> 5;
    const int lane = threadIdx.x & 31;
    const int token = blockIdx.x * 8 + wave;
    const int b = token >> 12;
    const int t = token & (TDIM - 1);
    const float* xb = x + (size_t)b * CDIM * TDIM;

    float h[16];
    float ss = 0.0f;
#pragma unroll
    for (int j = 0; j < 16; ++j) {
        const int c = lane + j * 32;
        const float* xc = xb + (size_t)c * TDIM;
        float acc = dwb[c];
#pragma unroll
        for (int k = 0; k < 7; ++k) {
            const int tt = t + k - 3;
            if (tt >= 0 && tt < TDIM) acc += xc[tt] * dww[c * 7 + k];
        }
        h[j] = acc;
        ss += acc * acc;
    }
    for (int o = 16; o > 0; o >>= 1) ss += __shfl_xor(ss, o, 32);
    const float r = rsqrtf(ss * (1.0f / (float)CDIM) + 1e-6f);

    float mx = 0.0f;
#pragma unroll
    for (int j = 0; j < 16; ++j) {
        const int c = lane + j * 32;
        h[j] = h[j] * r * nw[c];
        mx = fmaxf(mx, fabsf(h[j]));
    }
    for (int o = 16; o > 0; o >>= 1) mx = fmaxf(mx, __shfl_xor(mx, o, 32));
    const float sa = 127.0f / fmaxf(mx, 1e-5f);
#pragma unroll
    for (int j = 0; j < 16; ++j) {
        const int c = lane + j * 32;
        float q = fminf(fmaxf(rintf(h[j] * sa), -128.0f), 127.0f);
        a1[(size_t)token * CDIM + c] = (int8_t)q;
    }
    if (lane == 0) inv_sa1[token] = 1.0f / sa;
}

__device__ __forceinline__ v8i load_a_frag(const int8_t* As, int ldk, int mrow,
                                           int khalf, int ks) {
    const uint32_t* arow = (const uint32_t*)(As + (size_t)mrow * ldk + ks);
    v8i f;
#pragma unroll
    for (int j = 0; j < 8; ++j) {
        const int kb = ((j >> 2) * 32) + ((j & 3) * 4) + khalf * 2;
        f[j] = (int)arow[kb >> 2];
    }
    return f;
}

__device__ __forceinline__ v8i load_b_frag(const int8_t* Bw, int ldk, int nrow,
                                           int koffB, int ks) {
    const uint32_t* brow = (const uint32_t*)(Bw + (size_t)nrow * ldk + ks);
    v8i f;
#pragma unroll
    for (int j = 0; j < 4; ++j) {
        f[j]     = (int)brow[(koffB >> 2) + j];
        f[4 + j] = (int)brow[((32 + koffB) >> 2) + j];
    }
    return f;
}

__global__ __launch_bounds__(256, 1) void gemm1_kernel(
    const int8_t* __restrict__ A, const int8_t* __restrict__ Bw,
    const float* __restrict__ inv_sa, const float* __restrict__ scal,
    const float* __restrict__ b1, const float* __restrict__ la,
    const float* __restrict__ lb, float* __restrict__ H) {
    __shared__ int8_t As[32 * CDIM];
    const int m0 = blockIdx.x * 32;
    const int n0 = blockIdx.y * 512;
    async_tile_load(A + (size_t)m0 * CDIM, As, 32 * CDIM, threadIdx.x, 256);
    async_wait_all();
    __syncthreads();

    const int wave  = threadIdx.x >> 5;
    const int lane  = threadIdx.x & 31;
    const int msub  = wave >> 2;
    const int mrow  = msub * 16 + (lane & 15);
    const int khalf = (lane >> 4) * 8;
    const int ncol  = lane & 15;
    const int koffB = (lane >> 4) * 16;
    const int nbase = n0 + (wave & 3) * 128;

    v8i acc[8];
#pragma unroll
    for (int s = 0; s < 8; ++s)
#pragma unroll
        for (int r = 0; r < 8; ++r) acc[s][r] = 0;

#pragma unroll 1
    for (int ks = 0; ks < CDIM; ks += 64) {
        const v8i afrag = load_a_frag(As, CDIM, mrow, khalf, ks);
#pragma unroll
        for (int s = 0; s < 8; ++s) {
            const v8i bfrag = load_b_frag(Bw, CDIM, nbase + s * 16 + ncol, koffB, ks);
            acc[s] = __builtin_amdgcn_wmma_i32_16x16x64_iu8(
                true, afrag, true, bfrag, acc[s], false, false);
        }
    }

    const float inv_sw = fmaxf(scal[0] * (1.0f / (float)(IDIM * CDIM)), 1e-5f);
#pragma unroll
    for (int s = 0; s < 8; ++s) {
        const int n = nbase + s * 16 + ncol;
        const float bias = b1[n];
        const float a = __expf(la[n]);
        const float ib = 1.0f / (__expf(lb[n]) + 1e-9f);
#pragma unroll
        for (int r = 0; r < 8; ++r) {
            const int token = m0 + msub * 16 + r + ((lane >> 4) << 3);
            float y = (float)acc[s][r] * inv_sa[token] * inv_sw + bias;
            const float sn = __sinf(a * y);
            H[(size_t)token * IDIM + n] = y + ib * sn * sn;
        }
    }
}

__global__ __launch_bounds__(256) void quant2_kernel(const float* __restrict__ H,
                                                     int8_t* __restrict__ a2,
                                                     float* __restrict__ inv_sa2) {
    __shared__ float red[8];
    const int token = blockIdx.x;
    const float* hrow = H + (size_t)token * IDIM;
    float mx = 0.0f;
    for (int i = threadIdx.x; i < IDIM; i += 256) mx = fmaxf(mx, fabsf(hrow[i]));
    for (int o = 16; o > 0; o >>= 1) mx = fmaxf(mx, __shfl_xor(mx, o, 32));
    if ((threadIdx.x & 31) == 0) red[threadIdx.x >> 5] = mx;
    __syncthreads();
    if (threadIdx.x == 0) {
        float m = red[0];
#pragma unroll
        for (int i = 1; i < 8; ++i) m = fmaxf(m, red[i]);
        red[0] = m;
    }
    __syncthreads();
    mx = red[0];
    const float sa = 127.0f / fmaxf(mx, 1e-5f);
    for (int i = threadIdx.x; i < IDIM; i += 256) {
        float q = fminf(fmaxf(rintf(hrow[i] * sa), -128.0f), 127.0f);
        a2[(size_t)token * IDIM + i] = (int8_t)q;
    }
    if (threadIdx.x == 0) inv_sa2[token] = 1.0f / sa;
}

__global__ __launch_bounds__(128, 1) void gemm2_kernel(
    const int8_t* __restrict__ A, const int8_t* __restrict__ Bw,
    const float* __restrict__ inv_sa, const float* __restrict__ scal,
    const float* __restrict__ b2, const float* __restrict__ gamma,
    const float* __restrict__ x, float* __restrict__ out) {
    __shared__ int8_t As[16 * IDIM];
    const int m0 = blockIdx.x * 16;
    async_tile_load(A + (size_t)m0 * IDIM, As, 16 * IDIM, threadIdx.x, 128);
    async_wait_all();
    __syncthreads();

    const int wave  = threadIdx.x >> 5;
    const int lane  = threadIdx.x & 31;
    const int mrow  = lane & 15;
    const int khalf = (lane >> 4) * 8;
    const int ncol  = lane & 15;
    const int koffB = (lane >> 4) * 16;
    const int nbase = wave * 128;

    v8i acc[8];
#pragma unroll
    for (int s = 0; s < 8; ++s)
#pragma unroll
        for (int r = 0; r < 8; ++r) acc[s][r] = 0;

#pragma unroll 1
    for (int ks = 0; ks < IDIM; ks += 64) {
        const v8i afrag = load_a_frag(As, IDIM, mrow, khalf, ks);
#pragma unroll
        for (int s = 0; s < 8; ++s) {
            const v8i bfrag = load_b_frag(Bw, IDIM, nbase + s * 16 + ncol, koffB, ks);
            acc[s] = __builtin_amdgcn_wmma_i32_16x16x64_iu8(
                true, afrag, true, bfrag, acc[s], false, false);
        }
    }

    const float inv_sw = fmaxf(scal[1] * (1.0f / (float)(IDIM * CDIM)), 1e-5f);
#pragma unroll
    for (int s = 0; s < 8; ++s) {
        const int n = nbase + s * 16 + ncol;
        const float bias = b2[n];
        const float g = gamma[n];
#pragma unroll
        for (int r = 0; r < 8; ++r) {
            const int token = m0 + r + ((lane >> 4) << 3);
            const int b = token >> 12;
            const int t = token & (TDIM - 1);
            const float y = (float)acc[s][r] * inv_sa[token] * inv_sw + bias;
            const size_t idx = (size_t)b * CDIM * TDIM + (size_t)n * TDIM + t;
            out[idx] = x[idx] + g * y;
        }
    }
}

extern "C" void kernel_launch(void* const* d_in, const int* in_sizes, int n_in,
                              void* d_out, int out_size, void* d_ws, size_t ws_size,
                              hipStream_t stream) {
    const float* x     = (const float*)d_in[0];
    const float* dw_w  = (const float*)d_in[1];
    const float* dw_b  = (const float*)d_in[2];
    const float* nrm_w = (const float*)d_in[3];
    const float* w1    = (const float*)d_in[4];
    const float* b1    = (const float*)d_in[5];
    const float* la1   = (const float*)d_in[6];
    const float* lb1   = (const float*)d_in[7];
    const float* w2    = (const float*)d_in[8];
    const float* b2    = (const float*)d_in[9];
    const float* gam   = (const float*)d_in[10];
    float* out = (float*)d_out;

    char* ws = (char*)d_ws;
    float*  scal    = (float*)ws;
    int8_t* w1q     = (int8_t*)(ws + 256);
    int8_t* w2q     = w1q + (size_t)IDIM * CDIM;
    int8_t* a1      = w2q + (size_t)CDIM * IDIM;
    int8_t* a2      = a1 + (size_t)NTOK * CDIM;
    float*  inv_sa1 = (float*)(a2 + (size_t)NTOK * IDIM);
    float*  inv_sa2 = inv_sa1 + NTOK;
    float*  H       = inv_sa2 + NTOK;

    const int NW = IDIM * CDIM;

    init_scal_kernel<<<1, 64, 0, stream>>>(scal);
    absmean_kernel<<<256, 256, 0, stream>>>(w1, NW, scal + 0);
    absmean_kernel<<<256, 256, 0, stream>>>(w2, NW, scal + 1);
    wquant_kernel<<<1024, 256, 0, stream>>>(w1, NW, scal + 0, w1q);
    wquant_kernel<<<1024, 256, 0, stream>>>(w2, NW, scal + 1, w2q);

    conv_norm_quant_kernel<<<NTOK / 8, 256, 0, stream>>>(x, dw_w, dw_b, nrm_w,
                                                         a1, inv_sa1);

    gemm1_kernel<<<dim3(NTOK / 32, IDIM / 512), 256, 0, stream>>>(
        a1, w1q, inv_sa1, scal, b1, la1, lb1, H);

    quant2_kernel<<<NTOK, 256, 0, stream>>>(H, a2, inv_sa2);

    gemm2_kernel<<<NTOK / 16, 128, 0, stream>>>(a2, w2q, inv_sa2, scal, b2, gam,
                                                x, out);
}
